// Involution_29068338659813
// MI455X (gfx1250) — compile-verified, matrix operand order checked by reading, not run on hardware
//
#include <hip/hip_runtime.h>
#include <math.h>

typedef __attribute__((ext_vector_type(16))) _Float16 v16h;
typedef __attribute__((ext_vector_type(8)))  float    v8f;

// CDNA5 async global->LDS path (guarded: falls back to VGPR staging if absent)
#if defined(__has_builtin)
# if __has_builtin(__builtin_amdgcn_global_load_async_to_lds_b32) && \
     __has_builtin(__builtin_amdgcn_s_wait_asynccnt)
#  define USE_ASYNC_LDS 1
# endif
#endif
#ifndef USE_ASYNC_LDS
# define USE_ASYNC_LDS 0
#endif

#if USE_ASYNC_LDS
typedef __attribute__((address_space(1))) int gas_int;   // global (addrspace 1)
typedef __attribute__((address_space(3))) int las_int;   // LDS    (addrspace 3)
#endif

// Problem constants (fixed by the reference)
constexpr int BB   = 4;
constexpr int CCH  = 64;    // channels
constexpr int RCH  = 16;    // reduced channels
constexpr int GRP  = 16;    // groups
constexpr int KT   = 7;     // kernel taps per dim
constexpr int PAD  = KT / 2;          // 3
constexpr int TAPS = KT * KT;         // 49
constexpr int OCH  = TAPS * GRP;      // 784 span outputs
constexpr int HH   = 128;
constexpr int WW   = 128;
constexpr int TILE = 16;              // spatial tile (16x16 pixels per workgroup)
constexpr int PADW = TILE + KT - 1;   // 22
constexpr int XSROW = PADW * PADW;    // 484 elements per channel in LDS
constexpr size_t SMEM_BYTES = (size_t)CCH * XSROW * sizeof(float)   // 123904: padded x tile, f32
                            + (size_t)OCH * RCH * sizeof(_Float16); // 25088 : Ws in f16
// total 148992 B -> 2 workgroups per 320KB WGP

__global__ __launch_bounds__(256)
void involution_fused(const float* __restrict__ x,
                      const float* __restrict__ wr, const float* __restrict__ br,
                      const float* __restrict__ ws, const float* __restrict__ bs,
                      float* __restrict__ out)
{
    extern __shared__ char smem[];
    float*    xs  = (float*)smem;                                  // [64][484] padded x tile
    _Float16* wsl = (_Float16*)(smem + (size_t)CCH * XSROW * 4);   // [784][16] span weights f16

    const int tid  = threadIdx.x;
    const int lane = tid & 31;
    const int wave = tid >> 5;
    const int w0   = blockIdx.x * TILE;
    const int h0   = blockIdx.y * TILE;
    const int b    = blockIdx.z;

    // ---- stage: padded x tile (f32, zero halo) + span weights (f16) into LDS ----
    const float* xb = x + (size_t)b * CCH * HH * WW;
#if USE_ASYNC_LDS
    const bool interior = (h0 >= PAD) && (h0 + TILE + PAD <= HH) &&
                          (w0 >= PAD) && (w0 + TILE + PAD <= WW);
    if (interior) {
        // memory -> LDS directly (GLOBAL_LOAD_ASYNC_TO_LDS_B32, ASYNCcnt-tracked):
        // no VGPR staging, overlaps with the Ws f16 conversion below.
        for (int c = 0; c < CCH; ++c) {
            const float* xc = xb + (size_t)c * HH * WW + (size_t)(h0 - PAD) * WW + (w0 - PAD);
            float* xsc = xs + c * XSROW;
            for (int p = tid; p < XSROW; p += 256) {
                int r = p / PADW, q = p - r * PADW;
                __builtin_amdgcn_global_load_async_to_lds_b32(
                    (gas_int*)(xc + r * WW + q),
                    (las_int*)(xsc + p),
                    0, 0);
            }
        }
    } else
#endif
    {
        for (int c = 0; c < CCH; ++c) {
            const float* xc = xb + (size_t)c * HH * WW;
            for (int p = tid; p < XSROW; p += 256) {
                int r = p / PADW, q = p - r * PADW;
                int gh = h0 - PAD + r, gw = w0 - PAD + q;
                float v = 0.f;
                if (gh >= 0 && gh < HH && gw >= 0 && gw < WW) v = xc[gh * WW + gw];
                xs[c * XSROW + p] = v;
            }
        }
    }
    for (int i = tid; i < OCH * RCH; i += 256) wsl[i] = (_Float16)ws[i];
#if USE_ASYNC_LDS
    __builtin_amdgcn_s_wait_asynccnt(0);   // this wave's async->LDS done before barrier
#endif
    __syncthreads();

    // Fragment geometry (wave32 WMMA 16x16):
    //   A (16x32 f16): lane m & m+16 hold row M=m&15; K sub-base kb = lane<16 ? 0 : 8
    //                  element e -> K = kb+e (e<8) | kb+e+8 (e>=8)
    //   B (32x16 f16): same K pattern, lane -> column N = lane&15
    //   C/D (16x16 f32): VGPR r -> M = r + kb, lane -> N = lane&15
    const int kb = (lane < 16) ? 0 : 8;
    const int nr = lane & 15;

    // ---- A fragments for reduce GEMM: Wr is 16x64 row-major ----
    v16h a_r0, a_r1;
    #pragma unroll
    for (int e = 0; e < 16; ++e) {
        int k = (e < 8) ? (kb + e) : (kb + e + 8);
        a_r0[e] = (_Float16)wr[nr * CCH + k];
        a_r1[e] = (_Float16)wr[nr * CCH + k + 32];
    }

    // ---- reduce GEMM: red = relu(Wr @ x + br), one 16-pixel row per WMMA N-tile ----
    // D-layout of red IS the B-layout of the span GEMM (K zero-padded 16->32): keep in regs.
    v16h redB[2];
    #pragma unroll
    for (int jj = 0; jj < 2; ++jj) {
        const int jr   = wave * 2 + jj;                      // row of the 16x16 tile
        const int base = (jr + PAD) * PADW + (nr + PAD);     // center pixel in xs
        v16h b0, b1;
        #pragma unroll
        for (int e = 0; e < 16; ++e) {
            int k = (e < 8) ? (kb + e) : (kb + e + 8);
            b0[e] = (_Float16)xs[k * XSROW + base];
            b1[e] = (_Float16)xs[(k + 32) * XSROW + base];
        }
        v8f acc = {};
        acc = __builtin_amdgcn_wmma_f32_16x16x32_f16(false, a_r0, false, b0, (short)0, acc, false, false);
        acc = __builtin_amdgcn_wmma_f32_16x16x32_f16(false, a_r1, false, b1, (short)0, acc, false, false);
        v16h rb = {};                                        // e>=8 stays 0 => K=16..31 padding
        #pragma unroll
        for (int r = 0; r < 8; ++r) {
            float v = acc[r] + br[r + kb];                   // acc[r] is red[M=r+kb][pixel nr]
            rb[r] = (_Float16)(v > 0.f ? v : 0.f);
        }
        redB[jj] = rb;
    }

    // ---- per group: span GEMM (4 M-tiles of 16 taps) -> softmax(49) -> 7x7 gather ----
    #pragma unroll 1
    for (int g = 0; g < GRP; ++g) {
        const int obase = g * TAPS;                          // rows g*49 .. g*49+48 of Ws
        v16h A[4];
        #pragma unroll
        for (int t = 0; t < 4; ++t) {
            v16h a = {};
            int row = obase + t * 16 + nr;
            if (row < OCH) {
                #pragma unroll
                for (int e = 0; e < 8; ++e) a[e] = wsl[row * RCH + kb + e];
            }
            A[t] = a;
        }
        #pragma unroll 1
        for (int jj = 0; jj < 2; ++jj) {
            const int jr = wave * 2 + jj;
            v8f d[4];
            #pragma unroll
            for (int t = 0; t < 4; ++t) {
                v8f z = {};
                d[t] = __builtin_amdgcn_wmma_f32_16x16x32_f16(false, A[t], false, redB[jj],
                                                              (short)0, z, false, false);
            }
            // bias + masked max over this lane's taps (tap = t*16 + kb + r)
            float mx = -3.0e38f;
            #pragma unroll
            for (int t = 0; t < 4; ++t)
                #pragma unroll
                for (int r = 0; r < 8; ++r) {
                    int tap = t * 16 + kb + r;
                    if (tap < TAPS) {
                        d[t][r] += bs[obase + tap];
                        mx = fmaxf(mx, d[t][r]);
                    }
                }
            // lanes n and n+16 hold complementary tap sets of pixel n
            mx = fmaxf(mx, __shfl_xor(mx, 16, 32));
            float sum = 0.f;
            #pragma unroll
            for (int t = 0; t < 4; ++t)
                #pragma unroll
                for (int r = 0; r < 8; ++r) {
                    int tap = t * 16 + kb + r;
                    float e = (tap < TAPS) ? __expf(d[t][r] - mx) : 0.f;
                    d[t][r] = e;
                    sum += e;
                }
            sum += __shfl_xor(sum, 16, 32);
            const float inv = 1.0f / sum;
            // involution: out[g*4+cc] = inv * sum_tap w_tap * xpad[h+ti][w+tj]
            #pragma unroll
            for (int cc = 0; cc < 4; ++cc) {
                const float* xrow = xs + (g * 4 + cc) * XSROW;
                float p = 0.f;
                #pragma unroll
                for (int t = 0; t < 4; ++t)
                    #pragma unroll
                    for (int r = 0; r < 8; ++r) {
                        int tap = t * 16 + kb + r;
                        if (tap < TAPS) {
                            int ti = tap / KT, tj = tap - ti * KT;
                            p += d[t][r] * xrow[(jr + ti) * PADW + (nr + tj)];
                        }
                    }
                p += __shfl_xor(p, 16, 32);                  // combine the two tap halves
                if (lane < 16)
                    out[((size_t)b * CCH + g * 4 + cc) * (HH * WW)
                        + (h0 + jr) * WW + (w0 + nr)] = p * inv;
            }
        }
    }
}

extern "C" void kernel_launch(void* const* d_in, const int* in_sizes, int n_in,
                              void* d_out, int out_size, void* d_ws, size_t ws_size,
                              hipStream_t stream) {
    const float* x  = (const float*)d_in[0];
    const float* wr = (const float*)d_in[1];
    const float* br = (const float*)d_in[2];
    const float* ws = (const float*)d_in[3];
    const float* bs = (const float*)d_in[4];
    float* out = (float*)d_out;

    (void)in_sizes; (void)n_in; (void)out_size; (void)d_ws; (void)ws_size;

    // 149 KB dynamic LDS > 64 KB default cap: raise the limit (host-side, capture-safe).
    (void)hipFuncSetAttribute((const void*)involution_fused,
                              hipFuncAttributeMaxDynamicSharedMemorySize, (int)SMEM_BYTES);

    dim3 grid(WW / TILE, HH / TILE, BB);   // 8 x 8 x 4 = 256 workgroups
    involution_fused<<<grid, dim3(256), SMEM_BYTES, stream>>>(x, wr, br, ws, bs, out);
}
